// InterpretableMultiHeadAttention_85950885527640
// MI455X (gfx1250) — matrix operand order repaired and checked by reading, not compiled, not run on hardware
//
#include <hip/hip_runtime.h>
#include <hip/hip_bf16.h>
#include <math.h>

// Dims from the reference
#define BB   4
#define LQ   1024
#define DDIM 1024
#define HH   16
#define DK   64

typedef __attribute__((ext_vector_type(16))) _Float16 v16h;
typedef __attribute__((ext_vector_type(8)))  _Float16 v8h;
typedef __attribute__((ext_vector_type(8)))  float    v8f;

// ---------------------------------------------------------------------------
// WMMA wrapper: D = A(16x32 f16) * B(32x16 f16) + C(16x16 f32)
// ---------------------------------------------------------------------------
__device__ __forceinline__ v8f wmma32(v16h a, v16h b, v8f c) {
  return __builtin_amdgcn_wmma_f32_16x16x32_f16(
      /*neg_a=*/false, a, /*neg_b=*/false, b,
      /*c_mod=*/(short)0, c, /*reuse_a=*/false, /*reuse_b=*/false);
}

// A-matrix fragment, 16x32 f16, row-major source with leading dim `ld`.
// ISA layout: lane m = lane&15; lanes 0-15 hold K 0-7 & 16-23, lanes 16-31
// hold K 8-15 & 24-31.  Two 128-bit loads per lane.
__device__ __forceinline__ v16h load_a(const _Float16* base, int ld, int lane) {
  const _Float16* p = base + (size_t)(lane & 15) * ld + ((lane < 16) ? 0 : 8);
  v8h lo = *(const v8h*)p;
  v8h hi = *(const v8h*)(p + 16);
  v16h a;
#pragma unroll
  for (int i = 0; i < 8; ++i) { a[i] = lo[i]; a[8 + i] = hi[i]; }
  return a;
}

// B-matrix fragment, 32x16 f16, where logical B[k][n] = src[n*ld + k]
// (B given as its transpose, row-major).  ISA layout: lane n = lane&15,
// lanes 0-15 hold K 0-7 & 16-23, lanes 16-31 hold K 8-15 & 24-31
// => per lane two 128-bit reads.
__device__ __forceinline__ v16h load_bt(const _Float16* base, int ld, int k0, int lane) {
  const _Float16* p = base + (size_t)(lane & 15) * ld + k0 + ((lane < 16) ? 0 : 8);
  v8h lo = *(const v8h*)p;
  v8h hi = *(const v8h*)(p + 16);
  v16h b;
#pragma unroll
  for (int i = 0; i < 8; ++i) { b[i] = lo[i]; b[8 + i] = hi[i]; }
  return b;
}

// ---------------------------------------------------------------------------
// Prepass kernels (all B-operands stored TRANSPOSED so GEMM B-frag loads are
// contiguous 128-bit per lane instead of 16 scattered u16 loads)
// ---------------------------------------------------------------------------
__global__ void k_cvt_f16(const float* __restrict__ src, _Float16* __restrict__ dst, int n) {
  int i = blockIdx.x * 256 + threadIdx.x;
  if (i < n) dst[i] = (_Float16)src[i];
}

// wq/wk [H, D, DK] fp32  ->  WT16 [N = H*DK, K = D] f16 (heads fused, transposed)
__global__ void k_permwT(const float* __restrict__ src, _Float16* __restrict__ dst) {
  int i = blockIdx.x * 256 + threadIdx.x;   // dst index, 0 .. H*D*DK-1 = 1M
  int d = i & 1023;                         // K index
  int n = i >> 10;                          // N index = h*64 + kk
  int h = n >> 6, kk = n & 63;
  dst[i] = (_Float16)src[((size_t)h * DDIM + d) * DK + kk];
}

// wv [D, DK] fp32 -> WvT16 [DK, D] f16
__global__ void k_cvtT_wv(const float* __restrict__ src, _Float16* __restrict__ dst) {
  int i = blockIdx.x * 256 + threadIdx.x;   // dst index, 0 .. DK*D-1 = 64K
  int d = i & 1023;
  int n = i >> 10;
  dst[i] = (_Float16)src[(size_t)d * DK + n];
}

__global__ void k_zero(float* __restrict__ p, int n) {
  int i = blockIdx.x * 256 + threadIdx.x;
  if (i < n) p[i] = 0.0f;
}

// ---------------------------------------------------------------------------
// Projection GEMM: Y = X16[4096,1024] * W (given as WT[N,1024]) + bias
// One wave computes a 16x64 strip (4 accumulators, A-frag reused 4x).
// Software-pipelined: next k-step fragments are in flight during WMMAs.
// grid(M/64, N/64), block 128 (4 waves cover 64 consecutive rows).
// storeT = 0: Y f16 row-major [M, N]
// storeT = 1: Y f16 stored per-batch transposed [B, N, L]  (for V path)
// ---------------------------------------------------------------------------
__global__ __launch_bounds__(128)
void k_gemm_proj(const _Float16* __restrict__ X, const _Float16* __restrict__ WT,
                 const float* __restrict__ bias, _Float16* __restrict__ Y,
                 int N, int storeT) {
  int wave = threadIdx.x >> 5, lane = threadIdx.x & 31;
  int mt   = blockIdx.x * 4 + wave;         // 16-row tile index
  int col0 = blockIdx.y * 64;
  const _Float16* Xb = X  + (size_t)mt * 16 * DDIM;
  const _Float16* Wb = WT + (size_t)col0 * DDIM;
  v8f c[4] = {{}, {}, {}, {}};

  v16h a = load_a(Xb, DDIM, lane);
  v16h b[4];
#pragma unroll
  for (int q = 0; q < 4; ++q) b[q] = load_bt(Wb + (size_t)(q * 16) * DDIM, DDIM, 0, lane);

#pragma unroll 2
  for (int ks = 0; ks < 31; ++ks) {
    // prefetch next k-step while current WMMAs consume the live fragments
    v16h an = load_a(Xb + (ks + 1) * 32, DDIM, lane);
    v16h bn[4];
#pragma unroll
    for (int q = 0; q < 4; ++q)
      bn[q] = load_bt(Wb + (size_t)(q * 16) * DDIM, DDIM, (ks + 1) * 32, lane);
#pragma unroll
    for (int q = 0; q < 4; ++q) c[q] = wmma32(a, b[q], c[q]);
    a = an;
#pragma unroll
    for (int q = 0; q < 4; ++q) b[q] = bn[q];
  }
#pragma unroll
  for (int q = 0; q < 4; ++q) c[q] = wmma32(a, b[q], c[q]);   // last k-step

  int mb = (lane < 16) ? 0 : 8;
  int nn = lane & 15;
#pragma unroll
  for (int q = 0; q < 4; ++q) {
    int col = col0 + q * 16 + nn;
    float bb = bias[col];
    if (!storeT) {
#pragma unroll
      for (int r = 0; r < 8; ++r)
        Y[(size_t)(mt * 16 + mb + r) * N + col] = (_Float16)(c[q][r] + bb);
    } else {
#pragma unroll
      for (int r = 0; r < 8; ++r) {
        int row = mt * 16 + mb + r;                 // global token row
        Y[((size_t)(row >> 10) * N + col) * LQ + (row & 1023)] =
            (_Float16)(c[q][r] + bb);
      }
    }
  }
}

// ---------------------------------------------------------------------------
// Fused attention: scores (WMMA) -> LDS -> softmax -> attn store (once) ->
// P*V (WMMA, A-frags from LDS, B-frags contiguous from transposed V) ->
// atomic head-mean accumulation.  grid(L/16, H, B), block 128 (4 waves).
// ---------------------------------------------------------------------------
__global__ __launch_bounds__(128)
void k_attn(const _Float16* __restrict__ Q, const _Float16* __restrict__ K,
            const _Float16* __restrict__ VT, float* __restrict__ attn,
            float* __restrict__ headacc) {
  __shared__ _Float16 P[16 * 1024];   // 32 KB score/prob tile
  __shared__ float red[128];
  __shared__ float rowmax[16];
  __shared__ float rinv[16];

  int qt = blockIdx.x, h = blockIdx.y, b = blockIdx.z;
  int wave = threadIdx.x >> 5, lane = threadIdx.x & 31, t = threadIdx.x;
  const float scale = 0.125f;         // 1/sqrt(64)

  // Q tile fragments (rows qt*16.., head slice h*64..), K-dim = 64 = 2 steps
  const _Float16* Qb = Q + ((size_t)(b * LQ + qt * 16)) * DDIM + h * DK;
  v16h a0 = load_a(Qb,      DDIM, lane);
  v16h a1 = load_a(Qb + 32, DDIM, lane);

  // S = (Q Kt) * scale -> LDS (f16).  Wave w covers n-tiles w, w+4, ...
  // Software-pipelined over the 16 n-tiles handled by this wave.
  const _Float16* Kbase = K + ((size_t)(b * LQ)) * DDIM + h * DK;
  {
    int n0 = wave * 16;
    v16h b0 = load_bt(Kbase + (size_t)n0 * DDIM, DDIM, 0, lane);
    v16h b1 = load_bt(Kbase + (size_t)n0 * DDIM, DDIM, 32, lane);
#pragma unroll 2
    for (int i = 0; i < 15; ++i) {
      int n0n = (wave + 4 * (i + 1)) * 16;
      v16h b0n = load_bt(Kbase + (size_t)n0n * DDIM, DDIM, 0, lane);
      v16h b1n = load_bt(Kbase + (size_t)n0n * DDIM, DDIM, 32, lane);
      v8f c = {};
      c = wmma32(a0, b0, c);
      c = wmma32(a1, b1, c);
      int mb = (lane < 16) ? 0 : 8;
      int col = n0 + (lane & 15);
#pragma unroll
      for (int r = 0; r < 8; ++r)
        P[(size_t)(mb + r) * 1024 + col] = (_Float16)(c[r] * scale);
      n0 = n0n; b0 = b0n; b1 = b1n;
    }
    v8f c = {};
    c = wmma32(a0, b0, c);
    c = wmma32(a1, b1, c);
    int mb = (lane < 16) ? 0 : 8;
    int col = n0 + (lane & 15);
#pragma unroll
    for (int r = 0; r < 8; ++r)
      P[(size_t)(mb + r) * 1024 + col] = (_Float16)(c[r] * scale);
  }
  __syncthreads();

  // row max (16 rows x 8 partials)
  int row = t >> 3, part = t & 7;
  float m = -1e30f;
  for (int j = 0; j < 128; ++j)
    m = fmaxf(m, (float)P[(size_t)row * 1024 + part * 128 + j]);
  red[t] = m;
  __syncthreads();
  if (part == 0) {
    float mm = red[t];
    for (int j = 1; j < 8; ++j) mm = fmaxf(mm, red[t + j]);
    rowmax[row] = mm;
  }
  __syncthreads();

  // exp + row sum; keep unnormalized e in LDS, fold 1/sum in later
  float rm = rowmax[row];
  float s = 0.0f;
  for (int j = 0; j < 128; ++j) {
    int c = part * 128 + j;
    float e = __expf((float)P[(size_t)row * 1024 + c] - rm);
    P[(size_t)row * 1024 + c] = (_Float16)e;
    s += e;
  }
  red[t] = s;
  __syncthreads();
  if (part == 0) {
    float ss = 0.0f;
    for (int j = 0; j < 8; ++j) ss += red[t + j];
    rinv[row] = 1.0f / ss;
  }
  __syncthreads();

  // single fp32 store of normalized attention [B, Lq, H, Lk] (coalesced)
  float* ab = attn + (((size_t)(b * LQ + qt * 16)) * HH + h) * LQ;
  for (int j = 0; j < 128; ++j) {
    int idx = t + 128 * j;
    int r = idx >> 10, c = idx & 1023;
    ab[(size_t)r * HH * LQ + c] = (float)P[idx] * rinv[r];
  }

  // heads tile = P[16,1024] @ V[1024,64]; wave w handles cols w*16..
  // VT layout [B, DK, L]: B[k][n] = VT[(b*DK + col0 + n)*L + k]  (contiguous)
  int col0 = wave * 16;
  const _Float16* Vb = VT + ((size_t)b * DK + col0) * LQ;
  v8f c2 = {};
  {
    v16h ap = load_a(&P[0], 1024, lane);
    v16h bv = load_bt(Vb, LQ, 0, lane);
#pragma unroll 2
    for (int ks = 0; ks < 31; ++ks) {
      v16h apn = load_a(&P[(ks + 1) * 32], 1024, lane);     // LDS A-frag
      v16h bvn = load_bt(Vb, LQ, (ks + 1) * 32, lane);      // global B-frag
      c2 = wmma32(ap, bv, c2);
      ap = apn; bv = bvn;
    }
    c2 = wmma32(ap, bv, c2);
  }
  int mb = (lane < 16) ? 0 : 8;
  int col = col0 + (lane & 15);
#pragma unroll
  for (int r = 0; r < 8; ++r)
    atomicAdd(&headacc[((size_t)(b * LQ) + qt * 16 + mb + r) * DK + col],
              c2[r] * rinv[mb + r] * (1.0f / (float)HH));
}

// ---------------------------------------------------------------------------
// out[m, :] = head_mean[m, :] @ wo[64,1024] + bo   (small: plain fp32)
// ---------------------------------------------------------------------------
__global__ __launch_bounds__(256)
void k_outproj(const float* __restrict__ hacc, const float* __restrict__ wo,
               const float* __restrict__ bo, float* __restrict__ out) {
  __shared__ float hrow[DK];
  int mrow = blockIdx.x;
  int t = threadIdx.x;
  if (t < DK) hrow[t] = hacc[(size_t)mrow * DK + t];
  __syncthreads();
#pragma unroll
  for (int cch = 0; cch < 4; ++cch) {
    int d = cch * 256 + t;
    float acc = bo[d];
#pragma unroll 8
    for (int k = 0; k < DK; ++k)
      acc = fmaf(hrow[k], wo[(size_t)k * DDIM + d], acc);
    out[(size_t)mrow * DDIM + d] = acc;
  }
}

// ---------------------------------------------------------------------------
extern "C" void kernel_launch(void* const* d_in, const int* in_sizes, int n_in,
                              void* d_out, int out_size, void* d_ws, size_t ws_size,
                              hipStream_t stream) {
  (void)in_sizes; (void)n_in; (void)out_size; (void)ws_size;
  const float* query  = (const float*)d_in[0];
  const float* key_in = (const float*)d_in[1];
  const float* value  = (const float*)d_in[2];
  const float* wq     = (const float*)d_in[3];
  const float* bq     = (const float*)d_in[4];
  const float* wk     = (const float*)d_in[5];
  const float* bk     = (const float*)d_in[6];
  const float* wv     = (const float*)d_in[7];
  const float* bv     = (const float*)d_in[8];
  const float* wo     = (const float*)d_in[9];
  const float* bo     = (const float*)d_in[10];

  const int M = BB * LQ;                 // 4096 token rows
  float* out_p  = (float*)d_out;                       // [B,L,D]
  float* attn_p = (float*)d_out + (size_t)M * DDIM;    // [B,L,H,L]

  // workspace carve-up (256B aligned), ~46 MiB total
  char* ws = (char*)d_ws; size_t off = 0;
  auto carve = [&](size_t bytes) { void* p = ws + off; off += (bytes + 255) & ~(size_t)255; return p; };
  _Float16* xq16  = (_Float16*)carve((size_t)M * DDIM * 2);
  _Float16* xk16  = (_Float16*)carve((size_t)M * DDIM * 2);
  _Float16* xv16  = (_Float16*)carve((size_t)M * DDIM * 2);
  _Float16* wqT16 = (_Float16*)carve((size_t)DDIM * DDIM * 2);
  _Float16* wkT16 = (_Float16*)carve((size_t)DDIM * DDIM * 2);
  _Float16* wvT16 = (_Float16*)carve((size_t)DK * DDIM * 2);
  _Float16* q16   = (_Float16*)carve((size_t)M * DDIM * 2);
  _Float16* k16   = (_Float16*)carve((size_t)M * DDIM * 2);
  _Float16* v16t  = (_Float16*)carve((size_t)M * DK * 2);   // [B, DK, L]
  float*    hacc  = (float*)   carve((size_t)M * DK * 4);

  // 1) fp32 -> f16 conversions / transposed weight layouts / zero accumulator
  int nTok = M * DDIM;                                   // 4,194,304
  k_cvt_f16<<<nTok / 256, 256, 0, stream>>>(query,  xq16, nTok);
  k_cvt_f16<<<nTok / 256, 256, 0, stream>>>(key_in, xk16, nTok);
  k_cvt_f16<<<nTok / 256, 256, 0, stream>>>(value,  xv16, nTok);
  k_permwT <<<(HH * DDIM * DK) / 256, 256, 0, stream>>>(wq, wqT16);
  k_permwT <<<(HH * DDIM * DK) / 256, 256, 0, stream>>>(wk, wkT16);
  k_cvtT_wv<<<(DDIM * DK) / 256, 256, 0, stream>>>(wv, wvT16);
  k_zero   <<<(M * DK) / 256, 256, 0, stream>>>(hacc, M * DK);

  // 2) projections (WMMA, 16x64 strip per wave, double-buffered fragments)
  k_gemm_proj<<<dim3(M / 64, DDIM / 64), 128, 0, stream>>>(xq16, wqT16, bq, q16, DDIM, 0);
  k_gemm_proj<<<dim3(M / 64, DDIM / 64), 128, 0, stream>>>(xk16, wkT16, bk, k16, DDIM, 0);
  k_gemm_proj<<<dim3(M / 64, 1),         128, 0, stream>>>(xv16, wvT16, bv, v16t, DK, 1);

  // 3) fused attention (WMMA scores + softmax-in-LDS + WMMA P*V)
  k_attn<<<dim3(LQ / 16, HH, BB), 128, 0, stream>>>(q16, k16, v16t, attn_p, hacc);

  // 4) output projection
  k_outproj<<<M, 256, 0, stream>>>(hacc, wo, bo, out_p);
}
